// BYOLTrainer_54915451847015
// MI455X (gfx1250) — compile-verified, matrix operand order checked by reading, not run on hardware
//
#include <hip/hip_runtime.h>
#include <hip/hip_bf16.h>
#include <math.h>

// ---------------------------------------------------------------------------
// BYOL trainer forward pass for MI455X (gfx1250), wave32 + WMMA f16.
// All GEMMs run through v_wmma_f32_16x16x32_f16 with zero-padded tiles:
//   d=100 -> Dp=128, T=100/102 -> Tp=128, head_dim 25 -> 32, 3d=300 -> 384.
// Weights are converted once per call to f16, transposed to [N, Kp] so both
// WMMA operands load as contiguous 8-half (16B) chunks per lane.
// Register blocking: each wave computes a 32x32 output patch (2x2 WMMA tiles,
// 4 f32 accumulators), so each K-step issues 4 back-to-back WMMAs per
// 8 b128 fragment loads (2x the FLOP/byte of 1-tile-per-wave).
// ---------------------------------------------------------------------------

typedef _Float16 half_t;
typedef __attribute__((ext_vector_type(16))) _Float16 v16h;
typedef __attribute__((ext_vector_type(8)))  _Float16 v8h;
typedef __attribute__((ext_vector_type(8)))  float    v8f;

#define ROI   100
#define TP    128
#define DP    128
#define NHEAD 4
#define HD    25
#define HDP   32
#define DFFN  1024
#define NB    512
#define FD    512

// ------------------------------ WMMA GEMM ----------------------------------
template<int ACT>
__device__ inline void store_tile(const v8f& acc, const float* __restrict__ bias,
                                  half_t* __restrict__ Cz, int ldc,
                                  int mt, int nt, int lane)
{
    const int m0 = mt * 16 + ((lane >> 4) << 3);
    const int n  = nt * 16 + (lane & 15);
    const float bv = bias ? bias[n] : 0.0f;
    half_t* cp = Cz + n;
#pragma unroll
    for (int r = 0; r < 8; ++r) {
        float v = acc[r] + bv;
        if (ACT == 1) v = v > 0.0f ? v : 0.0f;          // relu
        if (ACT == 2) v = v > 0.0f ? v : 0.01f * v;     // leaky_relu
        cp[(long long)(m0 + r) * ldc] = (half_t)v;
    }
}

// C[b, M, N] (f16) = act( A[b, M, Kp] (f16) x BT[b, N, Kp]^T (f16) + bias[N] )
// Kp multiple of 32; M, N multiples of 16. One wave per 32x32 output patch.
template<int ACT>
__global__ __launch_bounds__(128)
void wmma_gemm_k(const half_t* __restrict__ A, long long sA, int lda,
                 const half_t* __restrict__ BT, long long sB, int ldb,
                 const float* __restrict__ bias,
                 half_t* __restrict__ C, long long sC, int ldc,
                 int M, int N, int Kp)
{
    const int lane = threadIdx.x & 31;
    const int wv   = threadIdx.x >> 5;           // 4 waves / block, 2x2 layout
    const int nt0  = blockIdx.x * 4 + (wv & 1) * 2;
    const int mt0  = blockIdx.y * 4 + (wv >> 1) * 2;
    if (nt0 * 16 >= N || mt0 * 16 >= M) return;  // wave-uniform; EXEC stays full
    const int ntiles = N >> 4, mtiles = M >> 4;
    const bool nv1 = (nt0 + 1 < ntiles);
    const bool mv1 = (mt0 + 1 < mtiles);
    const int nt1 = nv1 ? nt0 + 1 : nt0;         // clamp: loads stay in-bounds
    const int mt1 = mv1 ? mt0 + 1 : mt0;
    const int z = blockIdx.z;

    const half_t* Az = A  + (long long)z * sA;
    const half_t* Bz = BT + (long long)z * sB;
    const int ko = (lane >> 4) << 3;             // lanes 0-15: K+0, lanes 16-31: K+8
    const half_t* a0 = Az + (long long)(mt0 * 16 + (lane & 15)) * lda;
    const half_t* a1 = Az + (long long)(mt1 * 16 + (lane & 15)) * lda;
    const half_t* b0 = Bz + (long long)(nt0 * 16 + (lane & 15)) * ldb;
    const half_t* b1 = Bz + (long long)(nt1 * 16 + (lane & 15)) * ldb;

    v8f c00 = {}, c01 = {}, c10 = {}, c11 = {};
    for (int k = 0; k < Kp; k += 32) {
        union { v16h v; v8h h[2]; } ua0, ua1, ub0, ub1;
        ua0.h[0] = *(const v8h*)(a0 + k + ko);
        ua0.h[1] = *(const v8h*)(a0 + k + 16 + ko);
        ua1.h[0] = *(const v8h*)(a1 + k + ko);
        ua1.h[1] = *(const v8h*)(a1 + k + 16 + ko);
        ub0.h[0] = *(const v8h*)(b0 + k + ko);
        ub0.h[1] = *(const v8h*)(b0 + k + 16 + ko);
        ub1.h[0] = *(const v8h*)(b1 + k + ko);
        ub1.h[1] = *(const v8h*)(b1 + k + 16 + ko);
        if (k + 32 < Kp) {
            __builtin_prefetch(a0 + k + 32, 0, 1);   // global_prefetch_b8
            __builtin_prefetch(b0 + k + 32, 0, 1);
        }
        c00 = __builtin_amdgcn_wmma_f32_16x16x32_f16(false, ua0.v, false, ub0.v, (short)0, c00, false, false);
        c01 = __builtin_amdgcn_wmma_f32_16x16x32_f16(false, ua0.v, false, ub1.v, (short)0, c01, false, false);
        c10 = __builtin_amdgcn_wmma_f32_16x16x32_f16(false, ua1.v, false, ub0.v, (short)0, c10, false, false);
        c11 = __builtin_amdgcn_wmma_f32_16x16x32_f16(false, ua1.v, false, ub1.v, (short)0, c11, false, false);
    }

    half_t* Cz = C + (long long)z * sC;
    store_tile<ACT>(c00, bias, Cz, ldc, mt0, nt0, lane);
    if (nv1)        store_tile<ACT>(c01, bias, Cz, ldc, mt0, nt1, lane);
    if (mv1)        store_tile<ACT>(c10, bias, Cz, ldc, mt1, nt0, lane);
    if (mv1 && nv1) store_tile<ACT>(c11, bias, Cz, ldc, mt1, nt1, lane);
}

// --------------------------- weight conversion -----------------------------
// dstT[n, k] = src[k, n] (f32 row-major, leading dim srcld), zero padded.
__global__ void cvt_pad_T_k(const float* __restrict__ src, int srcld, int K, int N,
                            half_t* __restrict__ dst, int Kp, int Np)
{
    int i = blockIdx.x * blockDim.x + threadIdx.x;
    if (i >= Kp * Np) return;
    int k = i % Kp, n = i / Kp;
    dst[i] = (k < K && n < N) ? (half_t)src[(long long)k * srcld + n] : (half_t)0;
}

__global__ void pad_bias_k(const float* __restrict__ src, int N,
                           float* __restrict__ dst, int Np)
{
    int i = blockIdx.x * blockDim.x + threadIdx.x;
    if (i < Np) dst[i] = (i < N) ? src[i] : 0.0f;
}

__global__ void zero_k(float* __restrict__ p, int n)
{
    int i = blockIdx.x * blockDim.x + threadIdx.x;
    if (i < n) p[i] = 0.0f;
}

// ----------------------------- input prep ----------------------------------
__global__ void cvt_x_k(const float* __restrict__ BV, half_t* __restrict__ X)
{
    int i = blockIdx.x * blockDim.x + threadIdx.x;
    if (i >= NB * TP * DP) return;
    int c = i & 127, t = (i >> 7) & 127, b = i >> 14;
    X[i] = (c < ROI && t < ROI) ? (half_t)BV[((long long)b * ROI + t) * ROI + c]
                                : (half_t)0;
}

// cls/dist tokens + mask embedding + sinusoid positional table (T = 102).
__global__ void masked_x_k(const float* __restrict__ BV, const int* __restrict__ idx,
                           int nm, const float* __restrict__ me,
                           const float* __restrict__ ct, const float* __restrict__ dt,
                           half_t* __restrict__ X)
{
    int i = blockIdx.x * blockDim.x + threadIdx.x;
    if (i >= NB * TP * DP) return;
    int c = i & 127, t = (i >> 7) & 127, b = i >> 14;
    bool ok = (c < ROI) && (t < ROI + 2);
    float v = 0.0f;
    if (ok) {
        if (t == 0)      v = ct[c];
        else if (t == 1) v = dt[c];
        else {
            int tt = t - 2;
            bool msk = false;
            for (int q = 0; q < nm; ++q) msk |= (idx[b * nm + q] == tt);
            v = msk ? me[c] : BV[((long long)b * ROI + tt) * ROI + c];
        }
        float ang = (float)t * powf(10000.0f, -(float)(2 * (c >> 1)) * (1.0f / ROI));
        v += (c & 1) ? cosf(ang) : sinf(ang);
    }
    X[i] = ok ? (half_t)v : (half_t)0;
}

// ----------------------------- attention aux -------------------------------
// QKV[b, t, 384] -> Q,K [b*h, Tp, 32] (q scaled by 1/sqrt(25)), VT [b*h, 32, Tp]
__global__ void scatter_qkv_k(const half_t* __restrict__ QKV,
                              half_t* __restrict__ Q, half_t* __restrict__ K,
                              half_t* __restrict__ VT, int T)
{
    int i = blockIdx.x * blockDim.x + threadIdx.x;
    if (i >= NB * NHEAD * TP * HDP) return;
    int j = i & 31, t = (i >> 5) & 127, h = (i >> 12) & 3, b = i >> 14;
    bool ok = (j < HD) && (t < T);
    const half_t* s = QKV + ((long long)b * TP + t) * 384 + h * HD + j;
    half_t q = (half_t)0, k = (half_t)0, v = (half_t)0;
    if (ok) { q = (half_t)((float)s[0] * 0.2f); k = s[128]; v = s[256]; }
    long long hb = (long long)(b * NHEAD + h) * (TP * HDP);
    Q[hb + t * HDP + j]  = q;
    K[hb + t * HDP + j]  = k;
    VT[hb + j * TP + t]  = v;
}

__global__ void softmax_k(half_t* __restrict__ S, int T)
{
    int row = blockIdx.x;                 // NB*NHEAD*TP rows of length 128
    int t = threadIdx.x;
    half_t* p = S + (long long)row * TP;
    if ((row & 127) >= T) { p[t] = (half_t)0; return; }
    __shared__ float red[128];
    float v = (t < T) ? (float)p[t] : -1e30f;
    red[t] = v; __syncthreads();
    for (int s = 64; s > 0; s >>= 1) { if (t < s) red[t] = fmaxf(red[t], red[t + s]); __syncthreads(); }
    float mx = red[0]; __syncthreads();
    float e = (t < T) ? expf(v - mx) : 0.0f;
    red[t] = e; __syncthreads();
    for (int s = 64; s > 0; s >>= 1) { if (t < s) red[t] += red[t + s]; __syncthreads(); }
    p[t] = (half_t)(e / red[0]);
}

__global__ void merge_heads_k(const half_t* __restrict__ OH, half_t* __restrict__ O, int T)
{
    int i = blockIdx.x * blockDim.x + threadIdx.x;
    if (i >= NB * TP * DP) return;
    int c = i & 127, t = (i >> 7) & 127, b = i >> 14;
    half_t v = (half_t)0;
    if (c < ROI && t < T) {
        int h = c / HD, j = c % HD;
        v = OH[((long long)(b * NHEAD + h) * TP + t) * HDP + j];
    }
    O[i] = v;
}

// x = LayerNorm(x + y) over the real 100 features; padded rows/cols zeroed.
__global__ void resid_ln_k(half_t* __restrict__ X, const half_t* __restrict__ Y,
                           const float* __restrict__ w, const float* __restrict__ bb, int T)
{
    int row = blockIdx.x; int t = row & 127;
    half_t* x = X + (long long)row * DP;
    int c = threadIdx.x;
    if (t >= T) { x[c] = (half_t)0; return; }
    const half_t* y = Y + (long long)row * DP;
    float v = (c < ROI) ? (float)x[c] + (float)y[c] : 0.0f;
    __shared__ float red[128];
    red[c] = (c < ROI) ? v : 0.0f; __syncthreads();
    for (int s = 64; s > 0; s >>= 1) { if (c < s) red[c] += red[c + s]; __syncthreads(); }
    float mu = red[0] * (1.0f / ROI); __syncthreads();
    float d = (c < ROI) ? v - mu : 0.0f;
    red[c] = d * d; __syncthreads();
    for (int s = 64; s > 0; s >>= 1) { if (c < s) red[c] += red[c + s]; __syncthreads(); }
    float inv = rsqrtf(red[0] * (1.0f / ROI) + 1e-5f);
    x[c] = (c < ROI) ? (half_t)(d * inv * w[c] + bb[c]) : (half_t)0;
}

// ------------------------------- gathers -----------------------------------
__global__ void gather_g_k(const half_t* __restrict__ X, const int* __restrict__ idx,
                           int nm, half_t* __restrict__ G)
{
    int i = blockIdx.x * blockDim.x + threadIdx.x;
    if (i >= NB * nm * DP) return;
    int c = i & 127, r = i >> 7, m = r % nm, b = r / nm;
    int t = idx[b * nm + m] + 2;
    G[i] = X[((long long)b * TP + t) * DP + c];
}

__global__ void gather_nf_k(const half_t* __restrict__ D, half_t* __restrict__ NF)
{
    int i = blockIdx.x * blockDim.x + threadIdx.x;
    if (i >= NB * 800) return;
    int u = i % 8, t = (i / 8) % ROI, b = i / 800;
    NF[i] = D[((long long)b * TP + t) * 16 + u];
}

// ------------------------------- losses ------------------------------------
// Per-sample 10x10 contrastive scores, accuracy count + InfoNCE sum.
__global__ void mcl_stats_k(const float* __restrict__ BV, const int* __restrict__ idx,
                            const half_t* __restrict__ P, float* __restrict__ accs)
{
    __shared__ float E[10 * 100];
    __shared__ float Q[10 * 100];
    __shared__ float tot[100];
    int b = blockIdx.x;
    for (int i = threadIdx.x; i < 1000; i += 128) {
        int m = i / 100, c = i % 100;
        E[i] = BV[((long long)b * ROI + idx[b * 10 + m]) * ROI + c];
        Q[i] = (float)P[(long long)(b * 10 + m) * DP + c];
    }
    __syncthreads();
    for (int p = threadIdx.x; p < 100; p += 128) {
        int m = p / 10, n = p % 10;
        float s = 0.0f;
        for (int c = 0; c < 100; ++c) s += E[m * 100 + c] * Q[n * 100 + c];
        tot[p] = s;
    }
    __syncthreads();
    if (threadIdx.x == 0) {
        float rmax[10], rlse[10], nce = 0.0f;
        for (int m = 0; m < 10; ++m) {
            float mx = -1e30f;
            for (int n = 0; n < 10; ++n) mx = fmaxf(mx, tot[m * 10 + n]);
            float se = 0.0f;
            for (int n = 0; n < 10; ++n) se += expf(tot[m * 10 + n] - mx);
            rmax[m] = mx; rlse[m] = logf(se);
            nce += tot[m * 10 + m] - mx - rlse[m];
        }
        int corr = 0;
        for (int n = 0; n < 10; ++n) {          // argmax over m of softmax rows
            float best = -1e30f; int am = 0;
            for (int m = 0; m < 10; ++m) {
                float sm = expf(tot[m * 10 + n] - rmax[m] - rlse[m]);
                if (sm > best) { best = sm; am = m; }
            }
            corr += (am == n);
        }
        atomicAdd(&accs[1], (float)corr);
        atomicAdd(&accs[2], nce);
    }
}

__global__ void mrm_mse_k(const float* __restrict__ BV, const int* __restrict__ idx,
                          const half_t* __restrict__ P, float* __restrict__ accs)
{
    int r = blockIdx.x; int m = r % 5, b = r / 5;
    int t = idx[b * 5 + m];
    int c = threadIdx.x;
    float d = 0.0f;
    if (c < ROI) {
        float p = (float)P[(long long)r * DP + c];
        float x = BV[((long long)b * ROI + t) * ROI + c];
        d = (p - x) * (p - x);
    }
    __shared__ float red[128];
    red[c] = d; __syncthreads();
    for (int s = 64; s > 0; s >>= 1) { if (c < s) red[c] += red[c + s]; __syncthreads(); }
    if (c == 0) atomicAdd(&accs[3], red[0]);
}

__global__ void reg_pair_k(const half_t* __restrict__ P, const half_t* __restrict__ Tg,
                           float* __restrict__ accs)
{
    int b = blockIdx.x; int t = threadIdx.x;
    float pp = 0.0f, tt = 0.0f, pt = 0.0f;
    for (int c = t; c < FD; c += 128) {
        float p = (float)P[(long long)b * FD + c];
        float q = (float)Tg[(long long)b * FD + c];
        pp += p * p; tt += q * q; pt += p * q;
    }
    __shared__ float r0[128], r1[128], r2[128];
    r0[t] = pp; r1[t] = tt; r2[t] = pt; __syncthreads();
    for (int s = 64; s > 0; s >>= 1) {
        if (t < s) { r0[t] += r0[t + s]; r1[t] += r1[t + s]; r2[t] += r2[t + s]; }
        __syncthreads();
    }
    if (t == 0) {
        float np_ = fmaxf(sqrtf(r0[0]), 1e-12f);
        float nt_ = fmaxf(sqrtf(r1[0]), 1e-12f);
        atomicAdd(&accs[0], 2.0f - 2.0f * r2[0] / (np_ * nt_));
    }
}

__global__ void finalize_k(const float* __restrict__ accs, float* __restrict__ out)
{
    if (threadIdx.x == 0) {
        out[0] = accs[0] / (float)NB;                 // loss
        out[1] = accs[1] / (float)(NB * 10);          // acc
        out[2] = accs[2] / (-(float)(NB * 10));       // nce
        out[3] = accs[3] / (float)(NB * 5 * ROI);     // mse
    }
}

// ------------------------------- host side ---------------------------------
struct EncW {
    half_t *qkvT, *outT, *ff1T, *ff2T, *drT, *fc1T, *fc2T;
    float  *pqkv_b, *pout_b, *pff2_b, *pdr_b;
    const float *ff1_b, *fc1_b, *fc2_b, *ln1_w, *ln1_b, *ln2_w, *ln2_b;
};

extern "C" void kernel_launch(void* const* d_in, const int* in_sizes, int n_in,
                              void* d_out, int out_size, void* d_ws, size_t ws_size,
                              hipStream_t stream)
{
    (void)in_sizes; (void)n_in; (void)out_size; (void)ws_size;
    auto F = [&](int i) { return (const float*)d_in[i]; };
    auto I = [&](int i) { return (const int*)d_in[i]; };

    const float* bv1 = F(0);
    const float* bv2 = F(1);
    const float* pred_w1  = F(38); const float* pred_b1  = F(39);
    const float* pred_w2  = F(40); const float* pred_b2  = F(41);
    const float* cpred_w1 = F(42); const float* cpred_b1 = F(43);
    const float* cpred_w2 = F(44); const float* cpred_b2 = F(45);
    const float* gpred_w1 = F(46); const float* gpred_b1 = F(47);
    const float* gpred_w2 = F(48); const float* gpred_b2 = F(49);
    const float* mask_e = F(50); const float* cls_t = F(51); const float* dist_t = F(52);
    const int* mcl_idx = I(53); const int* mrm_idx = I(54);

    // bump allocator over d_ws (256B aligned)
    char* wsb = (char*)d_ws; size_t off = 0;
    auto alloc = [&](size_t bytes) -> char* {
        off = (off + 255) & ~(size_t)255;
        char* p = wsb + off; off += bytes; return p;
    };

    auto cvtT = [&](const float* src, int srcld, int K, int N, half_t* dst, int Kp, int Np) {
        int tot = Kp * Np;
        cvt_pad_T_k<<<(tot + 255) / 256, 256, 0, stream>>>(src, srcld, K, N, dst, Kp, Np);
    };
    auto padb = [&](const float* src, int N, float* dst, int Np) {
        pad_bias_k<<<(Np + 255) / 256, 256, 0, stream>>>(src, N, dst, Np);
    };
    auto gemm = [&](int act, const half_t* A, long long sA, int lda,
                    const half_t* BT, long long sB, int ldb, const float* bias,
                    half_t* C, long long sC, int ldc, int M, int N, int K, int batch) {
        dim3 g((N + 63) / 64, (M + 63) / 64, batch), blk(128);
        if (act == 0)      wmma_gemm_k<0><<<g, blk, 0, stream>>>(A, sA, lda, BT, sB, ldb, bias, C, sC, ldc, M, N, K);
        else if (act == 1) wmma_gemm_k<1><<<g, blk, 0, stream>>>(A, sA, lda, BT, sB, ldb, bias, C, sC, ldc, M, N, K);
        else               wmma_gemm_k<2><<<g, blk, 0, stream>>>(A, sA, lda, BT, sB, ldb, bias, C, sC, ldc, M, N, K);
    };

    // --- per-encoder weight conversion (f16 transposed + padded biases) ---
    auto make_enc = [&](int di) -> EncW {
        EncW w;
        w.qkvT  = (half_t*)alloc((size_t)4 * 384 * 128 * 2);
        w.outT  = (half_t*)alloc((size_t)4 * 128 * 128 * 2);
        w.ff1T  = (half_t*)alloc((size_t)4 * 1024 * 128 * 2);
        w.ff2T  = (half_t*)alloc((size_t)4 * 128 * 1024 * 2);
        w.drT   = (half_t*)alloc((size_t)16 * 128 * 2);
        w.fc1T  = (half_t*)alloc((size_t)256 * 800 * 2);
        w.fc2T  = (half_t*)alloc((size_t)512 * 256 * 2);
        w.pqkv_b = (float*)alloc((size_t)4 * 384 * 4);
        w.pout_b = (float*)alloc((size_t)4 * 128 * 4);
        w.pff2_b = (float*)alloc((size_t)4 * 128 * 4);
        w.pdr_b  = (float*)alloc((size_t)16 * 4);
        const float* qkv_w = F(di + 0);  const float* qkv_b = F(di + 1);
        const float* out_w = F(di + 2);  const float* out_b = F(di + 3);
        w.ln1_w = F(di + 4); w.ln1_b = F(di + 5);
        const float* ff1_w = F(di + 6);  w.ff1_b = F(di + 7);
        const float* ff2_w = F(di + 8);  const float* ff2_b = F(di + 9);
        w.ln2_w = F(di + 10); w.ln2_b = F(di + 11);
        const float* dr_w = F(di + 12);  const float* dr_b = F(di + 13);
        const float* fc1_w = F(di + 14); w.fc1_b = F(di + 15);
        const float* fc2_w = F(di + 16); w.fc2_b = F(di + 17);
        for (int l = 0; l < 4; ++l) {
            for (int blk = 0; blk < 3; ++blk) {   // q/k/v blocks of qkv_w [100,300]
                cvtT(qkv_w + l * 100 * 300 + blk * 100, 300, 100, 100,
                     w.qkvT + (size_t)l * 384 * 128 + (size_t)blk * 128 * 128, 128, 128);
                padb(qkv_b + l * 300 + blk * 100, 100, w.pqkv_b + l * 384 + blk * 128, 128);
            }
            cvtT(out_w + l * 100 * 100, 100, 100, 100, w.outT + (size_t)l * 128 * 128, 128, 128);
            padb(out_b + l * 100, 100, w.pout_b + l * 128, 128);
            cvtT(ff1_w + l * 100 * 1024, 1024, 100, 1024, w.ff1T + (size_t)l * 1024 * 128, 128, 1024);
            cvtT(ff2_w + l * 1024 * 100, 100, 1024, 100, w.ff2T + (size_t)l * 128 * 1024, 1024, 128);
            padb(ff2_b + l * 100, 100, w.pff2_b + l * 128, 128);
        }
        cvtT(dr_w, 8, 100, 8, w.drT, 128, 16);
        padb(dr_b, 8, w.pdr_b, 16);
        cvtT(fc1_w, 256, 800, 256, w.fc1T, 800, 256);
        cvtT(fc2_w, 512, 256, 512, w.fc2T, 256, 512);
        return w;
    };

    EncW won = make_enc(2);    // online
    EncW wtg = make_enc(20);   // target

    // heads
    half_t* predT1  = (half_t*)alloc((size_t)1024 * 512 * 2);
    half_t* predT2  = (half_t*)alloc((size_t)512 * 1024 * 2);
    half_t* cpredT1 = (half_t*)alloc((size_t)1024 * 128 * 2);
    half_t* cpredT2 = (half_t*)alloc((size_t)128 * 1024 * 2);
    half_t* gpredT1 = (half_t*)alloc((size_t)1024 * 128 * 2);
    half_t* gpredT2 = (half_t*)alloc((size_t)128 * 1024 * 2);
    float* pcpred_b2 = (float*)alloc(128 * 4);
    float* pgpred_b2 = (float*)alloc(128 * 4);
    cvtT(pred_w1, 1024, 512, 1024, predT1, 512, 1024);
    cvtT(pred_w2, 512, 1024, 512, predT2, 1024, 512);
    cvtT(cpred_w1, 1024, 100, 1024, cpredT1, 128, 1024);
    cvtT(cpred_w2, 100, 1024, 100, cpredT2, 1024, 128);
    padb(cpred_b2, 100, pcpred_b2, 128);
    cvtT(gpred_w1, 1024, 100, 1024, gpredT1, 128, 1024);
    cvtT(gpred_w2, 100, 1024, 100, gpredT2, 1024, 128);
    padb(gpred_b2, 100, pgpred_b2, 128);

    // activations
    half_t* xbuf   = (half_t*)alloc((size_t)NB * TP * DP * 2);
    half_t* qkvbuf = (half_t*)alloc((size_t)NB * TP * 384 * 2);
    half_t* qbuf   = (half_t*)alloc((size_t)NB * NHEAD * TP * HDP * 2);
    half_t* kbuf   = (half_t*)alloc((size_t)NB * NHEAD * TP * HDP * 2);
    half_t* vtbuf  = (half_t*)alloc((size_t)NB * NHEAD * HDP * TP * 2);
    half_t* ohbuf  = (half_t*)alloc((size_t)NB * NHEAD * TP * HDP * 2);
    half_t* obuf   = (half_t*)alloc((size_t)NB * TP * DP * 2);
    half_t* ybuf   = (half_t*)alloc((size_t)NB * TP * DP * 2);
    half_t* hbuf   = (half_t*)alloc((size_t)NB * TP * DFFN * 2);   // FF hidden
    half_t* scores = hbuf;                                         // alias (disjoint lifetime)
    half_t* drout  = (half_t*)alloc((size_t)NB * TP * 16 * 2);
    half_t* nfbuf  = (half_t*)alloc((size_t)NB * 800 * 2);
    half_t* z1buf  = (half_t*)alloc((size_t)NB * 256 * 2);
    half_t* zo1    = (half_t*)alloc((size_t)NB * FD * 2);
    half_t* zo2    = (half_t*)alloc((size_t)NB * FD * 2);
    half_t* zt1    = (half_t*)alloc((size_t)NB * FD * 2);
    half_t* zt2    = (half_t*)alloc((size_t)NB * FD * 2);
    half_t* p1b    = (half_t*)alloc((size_t)NB * FD * 2);
    half_t* p2b    = (half_t*)alloc((size_t)NB * FD * 2);
    half_t* phbuf  = (half_t*)alloc((size_t)NB * 1024 * 2);
    half_t* gbuf   = (half_t*)alloc((size_t)NB * 10 * DP * 2);
    half_t* chbuf  = (half_t*)alloc((size_t)NB * 10 * 1024 * 2);
    half_t* predbf = (half_t*)alloc((size_t)NB * 10 * DP * 2);
    float*  accs   = (float*)alloc(4 * 4);

    zero_k<<<1, 32, 0, stream>>>(accs, 4);

    const int EW = NB * TP * DP;           // elementwise total over padded x

    auto run_encoder = [&](const EncW& w, int T) {
        for (int l = 0; l < 4; ++l) {
            gemm(0, xbuf, TP * DP, DP, w.qkvT + (size_t)l * 384 * 128, 0, 128,
                 w.pqkv_b + l * 384, qkvbuf, TP * 384, 384, TP, 384, DP, NB);
            scatter_qkv_k<<<(NB * NHEAD * TP * HDP + 255) / 256, 256, 0, stream>>>(
                qkvbuf, qbuf, kbuf, vtbuf, T);
            gemm(0, qbuf, TP * HDP, HDP, kbuf, TP * HDP, HDP, nullptr,
                 scores, TP * TP, TP, TP, TP, HDP, NB * NHEAD);
            softmax_k<<<NB * NHEAD * TP, 128, 0, stream>>>(scores, T);
            gemm(0, scores, TP * TP, TP, vtbuf, HDP * TP, TP, nullptr,
                 ohbuf, TP * HDP, HDP, TP, HDP, TP, NB * NHEAD);
            merge_heads_k<<<(EW + 255) / 256, 256, 0, stream>>>(ohbuf, obuf, T);
            gemm(0, obuf, TP * DP, DP, w.outT + (size_t)l * 128 * 128, 0, 128,
                 w.pout_b + l * 128, ybuf, TP * DP, DP, TP, DP, DP, NB);
            resid_ln_k<<<NB * TP, 128, 0, stream>>>(xbuf, ybuf, w.ln1_w + l * 100, w.ln1_b + l * 100, T);
            gemm(1, xbuf, TP * DP, DP, w.ff1T + (size_t)l * 1024 * 128, 0, 128,
                 w.ff1_b + l * 1024, hbuf, TP * DFFN, DFFN, TP, DFFN, DP, NB);
            gemm(0, hbuf, TP * DFFN, DFFN, w.ff2T + (size_t)l * 128 * 1024, 0, 1024,
                 w.pff2_b + l * 128, ybuf, TP * DP, DP, TP, DP, DFFN, NB);
            resid_ln_k<<<NB * TP, 128, 0, stream>>>(xbuf, ybuf, w.ln2_w + l * 100, w.ln2_b + l * 100, T);
        }
    };

    auto run_bntf = [&](const float* bv, const EncW& w, half_t* zout) {
        cvt_x_k<<<(EW + 255) / 256, 256, 0, stream>>>(bv, xbuf);
        run_encoder(w, 100);
        gemm(2, xbuf, 0, DP, w.drT, 0, 128, w.pdr_b, drout, 0, 16, NB * TP, 16, DP, 1);
        gather_nf_k<<<(NB * 800 + 255) / 256, 256, 0, stream>>>(drout, nfbuf);
        gemm(2, nfbuf, 0, 800, w.fc1T, 0, 800, w.fc1_b, z1buf, 0, 256, NB, 256, 800, 1);
        gemm(0, z1buf, 0, 256, w.fc2T, 0, 256, w.fc2_b, zout, 0, FD, NB, FD, 256, 1);
    };

    // ---- MCL branch (masked encode, T=102, online) ----
    masked_x_k<<<(EW + 255) / 256, 256, 0, stream>>>(bv1, mcl_idx, 10, mask_e, cls_t, dist_t, xbuf);
    run_encoder(won, 102);
    gather_g_k<<<(NB * 10 * DP + 255) / 256, 256, 0, stream>>>(xbuf, mcl_idx, 10, gbuf);
    gemm(2, gbuf, 0, DP, cpredT1, 0, 128, cpred_b1, chbuf, 0, 1024, NB * 10, 1024, DP, 1);
    gemm(0, chbuf, 0, 1024, cpredT2, 0, 1024, pcpred_b2, predbf, 0, DP, NB * 10, DP, 1024, 1);
    mcl_stats_k<<<NB, 128, 0, stream>>>(bv1, mcl_idx, predbf, accs);

    // ---- MRM branch (masked encode, T=102, online) ----
    masked_x_k<<<(EW + 255) / 256, 256, 0, stream>>>(bv1, mrm_idx, 5, mask_e, cls_t, dist_t, xbuf);
    run_encoder(won, 102);
    gather_g_k<<<(NB * 5 * DP + 255) / 256, 256, 0, stream>>>(xbuf, mrm_idx, 5, gbuf);
    gemm(2, gbuf, 0, DP, gpredT1, 0, 128, gpred_b1, chbuf, 0, 1024, NB * 5, 1024, DP, 1);
    gemm(0, chbuf, 0, 1024, gpredT2, 0, 1024, pgpred_b2, predbf, 0, DP, NB * 5, DP, 1024, 1);
    mrm_mse_k<<<NB * 5, 128, 0, stream>>>(bv1, mrm_idx, predbf, accs);

    // ---- BYOL feature passes ----
    run_bntf(bv1, won, zo1);
    run_bntf(bv2, won, zo2);
    run_bntf(bv1, wtg, zt2);   // t2 = bntf(bv1, target)
    run_bntf(bv2, wtg, zt1);   // t1 = bntf(bv2, target)

    // predictor MLP on online features
    gemm(1, zo1, 0, FD, predT1, 0, 512, pred_b1, phbuf, 0, 1024, NB, 1024, FD, 1);
    gemm(0, phbuf, 0, 1024, predT2, 0, 1024, pred_b2, p1b, 0, FD, NB, FD, 1024, 1);
    gemm(1, zo2, 0, FD, predT1, 0, 512, pred_b1, phbuf, 0, 1024, NB, 1024, FD, 1);
    gemm(0, phbuf, 0, 1024, predT2, 0, 1024, pred_b2, p2b, 0, FD, NB, FD, 1024, 1);

    reg_pair_k<<<NB, 128, 0, stream>>>(p1b, zt1, accs);
    reg_pair_k<<<NB, 128, 0, stream>>>(p2b, zt2, accs);

    finalize_k<<<1, 32, 0, stream>>>(accs, (float*)d_out);
}
